// SelfAttentionLayer_87136296501908
// MI455X (gfx1250) — matrix operand order repaired and checked by reading, not compiled, not run on hardware
//
#include <hip/hip_runtime.h>

// Self-attention for MI455X (gfx1250, wave32, WMMA + async global->LDS).
// Pipeline: cvt X->bf16; transpose-cvt W; QKV GEMM; transpose V; S = Q K^T (bf16);
// in-place softmax; out = P V (f32). All GEMMs: v_wmma_f32_16x16x32_bf16,
// staging via global_load_async_to_lds_b128, double-buffered 64-deep K-tiles
// (16 WMMAs per barrier).

typedef __bf16 bf16_t;
typedef __attribute__((ext_vector_type(16))) __bf16 v16bf;
typedef __attribute__((ext_vector_type(8)))  __bf16 v8bf;
typedef __attribute__((ext_vector_type(8)))  float  v8f;

#define KT   64   // K-tile depth (2 WMMA k-steps per barrier)
#define LDSS 72   // padded LDS row stride (elements): conflict-free 16B fragment reads

// low 32 bits of a flat shared pointer == wave-relative LDS byte offset
__device__ __forceinline__ unsigned lds_addr32(const void* p) {
  return (unsigned)(uintptr_t)p;
}

// CDNA5 async copy: LDS[vdst] = MEM[saddr + vaddr] (16B/lane); tracked by ASYNCcnt.
__device__ __forceinline__ void async_load_b128(unsigned lds_byte_off,
                                                const void* sbase,
                                                unsigned vbyte_off) {
  asm volatile("global_load_async_to_lds_b128 %0, %1, %2"
               :: "v"(lds_byte_off), "v"(vbyte_off), "s"(sbase)
               : "memory");
}

__device__ __forceinline__ void wait_async0() {
  asm volatile("s_wait_asynccnt 0x0" ::: "memory");
}

// A fragment (16x32 bf16, contiguous-k order): lane m=lane&15, half hk=lane>>4.
// elements 0..15 = K = hk*16 + j (contiguous, matches B fragment).
__device__ __forceinline__ v16bf load_a_frag(const bf16_t* rowbase) {
  const int lane = threadIdx.x & 31;
  const int m = lane & 15, hk = lane >> 4;
  const bf16_t* p = rowbase + m * LDSS;
  v8bf lo = *(const v8bf*)(p + hk * 16);
  v8bf hi = *(const v8bf*)(p + hk * 16 + 8);
  v16bf r;
#pragma unroll
  for (int i = 0; i < 8; ++i) { r[i] = lo[i]; r[i + 8] = hi[i]; }
  return r;
}

// B fragment (32x16 bf16): lane n=lane&15 holds K = (lane>>4)*16 + j, j=0..15.
// LDS tile holds B^T rows (i.e. [n][k]) -> 16 contiguous elements per lane.
__device__ __forceinline__ v16bf load_b_frag(const bf16_t* colbase) {
  const int lane = threadIdx.x & 31;
  const int n = lane & 15, kb = (lane >> 4) * 16;
  const bf16_t* p = colbase + n * LDSS + kb;
  v8bf lo = *(const v8bf*)(p);
  v8bf hi = *(const v8bf*)(p + 8);
  v16bf r;
#pragma unroll
  for (int i = 0; i < 8; ++i) { r[i] = lo[i]; r[i + 8] = hi[i]; }
  return r;
}

// C = scale * (A @ B^T) per batch-z.  A: [M x K] row-major (lda); B: [N x K] row-major (ldb).
// Block: 256 thr = 8 waves, 128x128 C-tile; wave grid 2x4, wave tile 64x32 (4x2 WMMAs
// per 32-deep k-substep, 2 substeps per LDS tile). K-tiles of 64 double-buffered via
// async loads; one workgroup barrier + one asynccnt wait per 16 WMMAs.
template <bool OUT_F32>
__global__ __launch_bounds__(256)
void gemm_nt_bf16_wmma(const bf16_t* __restrict__ A, int lda, size_t aBatch,
                       const bf16_t* __restrict__ Bm, int ldb, size_t bBatch,
                       void* __restrict__ C, int ldc, size_t cBatch,
                       int K, float scale0, float scaleN)
{
  __shared__ __align__(16) bf16_t As[2][128 * LDSS];
  __shared__ __align__(16) bf16_t Bs[2][128 * LDSS];

  const int tid  = threadIdx.x;
  const int lane = tid & 31;
  const int wave = tid >> 5;
  const int wm = wave >> 2;        // 0..1
  const int wn = wave & 3;         // 0..3
  const int row0 = blockIdx.x * 128;
  const int col0 = blockIdx.y * 128;
  const int bz = blockIdx.z;
  const float scale = (bz == 0) ? scale0 : scaleN;

  const bf16_t* gA = A  + (size_t)bz * aBatch + (size_t)row0 * lda;
  const bf16_t* gB = Bm + (size_t)bz * bBatch + (size_t)col0 * ldb;
  const size_t cbo = (size_t)bz * cBatch;

  // per-thread staging geometry: four 16B chunks of each 128x64 tile (branch-free)
  unsigned vA[4], vB[4], dL[4];
#pragma unroll
  for (int i = 0; i < 4; ++i) {
    const int c = tid + i * 256;          // 0..1023
    const int r = c >> 3, e = (c & 7) * 8;
    vA[i] = (unsigned)((r * lda + e) * 2);
    vB[i] = (unsigned)((r * ldb + e) * 2);
    dL[i] = (unsigned)((r * LDSS + e) * 2);
  }
  const unsigned baseA[2] = { lds_addr32(&As[0][0]), lds_addr32(&As[1][0]) };
  const unsigned baseB[2] = { lds_addr32(&Bs[0][0]), lds_addr32(&Bs[1][0]) };

  auto issue_tile = [&](int t) {
    const int p = t & 1;
    const bf16_t* ga = gA + t * KT;
    const bf16_t* gb = gB + t * KT;
#pragma unroll
    for (int i = 0; i < 4; ++i) async_load_b128(baseA[p] + dL[i], ga, vA[i]);
#pragma unroll
    for (int i = 0; i < 4; ++i) async_load_b128(baseB[p] + dL[i], gb, vB[i]);
  };

  v8f acc[4][2] = {};
  const int nk = K / KT;

  issue_tile(0);
  for (int t = 0; t < nk; ++t) {
    wait_async0();          // my async writes for tile t have landed in LDS
    __syncthreads();        // everyone's have; everyone done reading buf[(t+1)&1] last round
    if (t + 1 < nk) issue_tile(t + 1);   // overlaps with WMMA below (other buffer)

    const bf16_t* Ab = As[t & 1];
    const bf16_t* Bb = Bs[t & 1];
#pragma unroll
    for (int s = 0; s < 2; ++s) {        // two 32-deep k-substeps per LDS tile
      v16bf af[4];
#pragma unroll
      for (int i = 0; i < 4; ++i)
        af[i] = load_a_frag(Ab + (wm * 64 + i * 16) * LDSS + s * 32);
#pragma unroll
      for (int j = 0; j < 2; ++j) {
        v16bf bfr = load_b_frag(Bb + (wn * 32 + j * 16) * LDSS + s * 32);
#pragma unroll
        for (int i = 0; i < 4; ++i)
          acc[i][j] = __builtin_amdgcn_wmma_f32_16x16x32_bf16(
              false, af[i], false, bfr, (short)0, acc[i][j], false, false);
      }
    }
  }

  // C/D layout: vgpr r -> row r + 8*(lane>=16); col = lane&15
  const int hk = lane >> 4, lc = lane & 15;
#pragma unroll
  for (int i = 0; i < 4; ++i) {
#pragma unroll
    for (int j = 0; j < 2; ++j) {
      const int rbase = row0 + wm * 64 + i * 16 + hk * 8;
      const int cc = col0 + wn * 32 + j * 16 + lc;
#pragma unroll
      for (int r = 0; r < 8; ++r) {
        const float v = acc[i][j][r] * scale;
        const size_t idx = cbo + (size_t)(rbase + r) * ldc + cc;
        if constexpr (OUT_F32) ((float*)C)[idx] = v;
        else                   ((bf16_t*)C)[idx] = (bf16_t)v;
      }
    }
  }
}

// In-place softmax over rows of 4096 bf16 (fp32 math). One 256-thread block per row.
__global__ __launch_bounds__(256)
void softmax_inplace_bf16(bf16_t* __restrict__ S)
{
  __shared__ float red[256];
  const int tid = threadIdx.x;
  bf16_t* row = S + (size_t)blockIdx.x * 4096;

  float v[16];
  float m = -3.4e38f;
#pragma unroll
  for (int i = 0; i < 16; ++i) {
    v[i] = (float)row[tid + i * 256];
    m = fmaxf(m, v[i]);
  }
  red[tid] = m; __syncthreads();
  for (int s = 128; s > 0; s >>= 1) {
    if (tid < s) red[tid] = fmaxf(red[tid], red[tid + s]);
    __syncthreads();
  }
  const float rmax = red[0];
  __syncthreads();

  float sum = 0.f;
#pragma unroll
  for (int i = 0; i < 16; ++i) { v[i] = __expf(v[i] - rmax); sum += v[i]; }
  red[tid] = sum; __syncthreads();
  for (int s = 128; s > 0; s >>= 1) {
    if (tid < s) red[tid] += red[tid + s];
    __syncthreads();
  }
  const float inv = 1.0f / red[0];
#pragma unroll
  for (int i = 0; i < 16; ++i) row[tid + i * 256] = (bf16_t)(v[i] * inv);
}

__global__ __launch_bounds__(256)
void cvt_f32_to_bf16(const float* __restrict__ in, bf16_t* __restrict__ out, int n)
{
  const int i = blockIdx.x * 256 + threadIdx.x;
  if (i < n) out[i] = (bf16_t)in[i];
}

// f32 in[R][C] -> bf16 out[C][R]   (32x32 LDS tile, 256 threads)
__global__ __launch_bounds__(256)
void transpose_f32_to_bf16(const float* __restrict__ in, bf16_t* __restrict__ out,
                           int R, int C)
{
  __shared__ float tile[32][33];
  const int tx = threadIdx.x & 31;
  const int ty = threadIdx.x >> 5;   // 0..7
  const int r0 = blockIdx.y * 32;
  const int c0 = blockIdx.x * 32;
#pragma unroll
  for (int i = 0; i < 4; ++i)
    tile[ty + i * 8][tx] = in[(size_t)(r0 + ty + i * 8) * C + (c0 + tx)];
  __syncthreads();
#pragma unroll
  for (int i = 0; i < 4; ++i)
    out[(size_t)(c0 + ty + i * 8) * R + (r0 + tx)] = (bf16_t)tile[tx][ty + i * 8];
}

// bf16 in[z][R][C] -> bf16 out[z][C][R]
__global__ __launch_bounds__(256)
void transpose_bf16_batched(const bf16_t* __restrict__ in, bf16_t* __restrict__ out,
                            int R, int C)
{
  __shared__ bf16_t tile[32][34];
  const size_t bo = (size_t)blockIdx.z * R * C;
  const int tx = threadIdx.x & 31;
  const int ty = threadIdx.x >> 5;
  const int r0 = blockIdx.y * 32;
  const int c0 = blockIdx.x * 32;
#pragma unroll
  for (int i = 0; i < 4; ++i)
    tile[ty + i * 8][tx] = in[bo + (size_t)(r0 + ty + i * 8) * C + (c0 + tx)];
  __syncthreads();
#pragma unroll
  for (int i = 0; i < 4; ++i)
    out[bo + (size_t)(c0 + ty + i * 8) * R + (r0 + tx)] = tile[tx][ty + i * 8];
}

extern "C" void kernel_launch(void* const* d_in, const int* in_sizes, int n_in,
                              void* d_out, int out_size, void* d_ws, size_t ws_size,
                              hipStream_t stream)
{
  (void)in_sizes; (void)n_in; (void)out_size; (void)ws_size;
  constexpr int Bn = 4, N = 4096, D = 1024, U = 1024;
  const size_t nX = (size_t)Bn * N * D;   // 16,777,216
  const size_t nW = (size_t)D * U;        //  1,048,576
  const size_t nQ = (size_t)Bn * N * U;   // 16,777,216

  const float* X  = (const float*)d_in[0];
  const float* WQ = (const float*)d_in[1];
  const float* WK = (const float*)d_in[2];
  const float* WV = (const float*)d_in[3];
  float* out = (float*)d_out;

  char* ws = (char*)d_ws;
  bf16_t* Xh  = (bf16_t*)ws;  ws += nX * sizeof(bf16_t);        // 32 MB
  bf16_t* WT  = (bf16_t*)ws;  ws += 3 * nW * sizeof(bf16_t);    //  6 MB (W^T: [U][D])
  bf16_t* QKV = (bf16_t*)ws;  ws += 3 * nQ * sizeof(bf16_t);    // 96 MB
  bf16_t* Vt  = (bf16_t*)ws;  ws += nQ * sizeof(bf16_t);        // 32 MB (V^T: [B][U][N])
  bf16_t* Smx = (bf16_t*)ws;                                    // 128 MB

  // 1) X -> bf16 ; W -> bf16 transposed ([U][D]) so all GEMMs are NT-form
  cvt_f32_to_bf16<<<dim3((unsigned)((nX + 255) / 256)), dim3(256), 0, stream>>>(X, Xh, (int)nX);
  transpose_f32_to_bf16<<<dim3(32, 32), dim3(256), 0, stream>>>(WQ, WT + 0 * nW, D, U);
  transpose_f32_to_bf16<<<dim3(32, 32), dim3(256), 0, stream>>>(WK, WT + 1 * nW, D, U);
  transpose_f32_to_bf16<<<dim3(32, 32), dim3(256), 0, stream>>>(WV, WT + 2 * nW, D, U);

  const float qscale = 1.0f / 32.0f;  // 1/sqrt(1024) folded into Q

  // 2) QKV = Xh @ W^T^T   (M=16384, N=1024, K=1024; z selects weight/output)
  gemm_nt_bf16_wmma<false><<<dim3(128, 8, 3), dim3(256), 0, stream>>>(
      Xh, D, 0,
      WT, D, nW,
      (void*)QKV, U, nQ,
      D, qscale, 1.0f);

  // 3) V [B][N][U] -> V^T [B][U][N] so out-GEMM is NT-form too
  transpose_bf16_batched<<<dim3(32, 128, 4), dim3(256), 0, stream>>>(
      QKV + 2 * nQ, Vt, N, U);

  // 4) S = Q @ K^T per batch  (M=N=4096, K=1024), bf16 scores
  gemm_nt_bf16_wmma<false><<<dim3(32, 32, 4), dim3(256), 0, stream>>>(
      QKV,      U, (size_t)N * U,
      QKV + nQ, U, (size_t)N * U,
      (void*)Smx, N, (size_t)N * N,
      U, 1.0f, 1.0f);

  // 5) row softmax in place (Bn*N rows)
  softmax_inplace_bf16<<<dim3(Bn * N), dim3(256), 0, stream>>>(Smx);

  // 6) out = P @ (V^T)^T per batch  (M=4096, N=1024, K=4096), f32 output
  gemm_nt_bf16_wmma<true><<<dim3(32, 8, 4), dim3(256), 0, stream>>>(
      Smx, N, (size_t)N * N,
      Vt,  N, (size_t)U * N,
      (void*)out, U, (size_t)N * U,
      N, 1.0f, 1.0f);
}
